// GPT_11458972746292
// MI455X (gfx1250) — compile-verified, matrix operand order checked by reading, not run on hardware
//
#include <hip/hip_runtime.h>
#include <hip/hip_bf16.h>
#include <math.h>
#include <stdint.h>

// babyGPT-50m dims
#define BN 2
#define TN 1024
#define VOCN 50257
#define CN 768
#define HN 12
#define NLAYER 6
#define DN 64
#define FFN (4 * CN)
#define MN (BN * TN)

typedef __attribute__((ext_vector_type(16))) __bf16 v16bf;
typedef __attribute__((ext_vector_type(8)))  float  v8f;

union FragA { v16bf v; unsigned u[8]; };

__device__ __forceinline__ unsigned short f32_to_bf16(float f) {
  union { float f; unsigned u; } x; x.f = f;
  unsigned r = x.u + 0x7FFFu + ((x.u >> 16) & 1u);   // round-to-nearest-even
  return (unsigned short)(r >> 16);
}

__device__ __forceinline__ v8f wmma_bf16(const FragA& a, const FragA& b, v8f c) {
  return __builtin_amdgcn_wmma_f32_16x16x32_bf16(
      /*neg_a=*/false, a.v, /*neg_b=*/false, b.v,
      /*c_mod=*/(short)0, c, /*reuse_a=*/false, /*reuse_b=*/false);
}

// ---------------------------------------------------------------------------
// fp32 -> bf16 weight conversion
// ---------------------------------------------------------------------------
__global__ __launch_bounds__(256) void cvt_k(const float* __restrict__ s,
                                             unsigned short* __restrict__ d, int n) {
  int i = blockIdx.x * 256 + threadIdx.x;
  if (i < n) d[i] = f32_to_bf16(s[i]);
}

// ---------------------------------------------------------------------------
// token + positional embedding  ->  x fp32 [B*T, C]
// ---------------------------------------------------------------------------
__global__ __launch_bounds__(256) void embed_k(const int* __restrict__ idx,
                                               const float* __restrict__ tok,
                                               const float* __restrict__ pos,
                                               float* __restrict__ X, int total) {
  int i = blockIdx.x * 256 + threadIdx.x;
  if (i >= total) return;
  int c  = i % CN;
  int bt = i / CN;
  int t  = bt % TN;
  X[i] = tok[(size_t)idx[bt] * CN + c] + pos[(size_t)t * CN + c];
}

// ---------------------------------------------------------------------------
// LayerNorm: fp32 row in -> bf16 row out (one row per block)
// ---------------------------------------------------------------------------
__global__ __launch_bounds__(256) void layernorm_k(const float* __restrict__ X,
                                                   const float* __restrict__ g,
                                                   const float* __restrict__ bb,
                                                   unsigned short* __restrict__ O,
                                                   int Cdim) {
  const int row = blockIdx.x;
  const float* xr = X + (size_t)row * Cdim;
  float s = 0.f, s2 = 0.f;
  for (int c = threadIdx.x; c < Cdim; c += 256) { float v = xr[c]; s += v; s2 += v * v; }
#pragma unroll
  for (int o = 16; o; o >>= 1) { s += __shfl_down(s, o, 32); s2 += __shfl_down(s2, o, 32); }
  __shared__ float rs[8], rs2[8];
  __shared__ float mean_s, rstd_s;
  if ((threadIdx.x & 31) == 0) { rs[threadIdx.x >> 5] = s; rs2[threadIdx.x >> 5] = s2; }
  __syncthreads();
  if (threadIdx.x == 0) {
    float t = 0.f, t2 = 0.f;
    for (int i = 0; i < 8; ++i) { t += rs[i]; t2 += rs2[i]; }
    float m = t / Cdim;
    mean_s = m;
    rstd_s = rsqrtf(t2 / Cdim - m * m + 1e-5f);
  }
  __syncthreads();
  float m = mean_s, r = rstd_s;
  unsigned short* orow = O + (size_t)row * Cdim;
  for (int c = threadIdx.x; c < Cdim; c += 256)
    orow[c] = f32_to_bf16((xr[c] - m) * r * g[c] + bb[c]);
}

// ---------------------------------------------------------------------------
// WMMA GEMM:  Out[M,N] = A_bf16[M,K] @ B_bf16[K,N] (+bias) (+GELU)
// OUT_MODE: 0 = f32 store, 1 = f32 residual add, 2 = bf16 store
// Block: 256 thr (8 waves), tile 64(M) x 128(N), BK=32.
// Register double-buffering: next K-tile's global loads issue before the
// current tile's 4-WMMA burst. B staged K-transposed so bf16 K-pairs are
// u32-contiguous for fragment ds_load_b128s.
// ---------------------------------------------------------------------------
template <int OUT_MODE, bool GELU_FLAG>
__global__ __launch_bounds__(256) void gemm_wmma_k(const unsigned short* __restrict__ A,
                                                   const unsigned short* __restrict__ Bm,
                                                   const float* __restrict__ bias,
                                                   void* __restrict__ Out,
                                                   int M, int N, int K) {
  __shared__ unsigned short As[64][32];     // [m][k]
  __shared__ unsigned short BsT[128][32];   // [n][k]
  const int tid  = threadIdx.x;
  const int lane = tid & 31;
  const int wave = tid >> 5;
  const int m0 = blockIdx.y * 64;
  const int n0 = blockIdx.x * 128;
  const int wrow = wave & 3;                // 16-row subtile
  const int wcol = wave >> 2;               // 0..1 -> 64-col group
  const int mloc = lane & 15;
  const int hh   = lane >> 4;

  v8f acc[4] = {v8f{}, v8f{}, v8f{}, v8f{}};

  unsigned       aR[4];
  unsigned short bLo[8], bHi[8];

  auto stageLoad = [&](int k0) {
#pragma unroll
    for (int i = 0; i < 4; ++i) {           // A tile 64x32 as u32 pairs
      int e = tid + i * 256;
      int r = e >> 4, cp = e & 15;
      int gr = m0 + r;
      aR[i] = (gr < M) ? *(const unsigned*)(A + (size_t)gr * K + k0 + cp * 2) : 0u;
    }
#pragma unroll
    for (int i = 0; i < 8; ++i) {           // B tile 32x128, u16 (N may be odd)
      int p = tid + i * 256;
      int kk = p >> 6, n = (p & 63) * 2;
      int gn = n0 + n;
      const unsigned short* src = Bm + (size_t)(k0 + kk) * N + gn;
      bLo[i] = (gn     < N) ? src[0] : (unsigned short)0;
      bHi[i] = (gn + 1 < N) ? src[1] : (unsigned short)0;
    }
  };
  auto stageStore = [&]() {
#pragma unroll
    for (int i = 0; i < 4; ++i) {
      int e = tid + i * 256;
      int r = e >> 4, cp = e & 15;
      *(unsigned*)&As[r][cp * 2] = aR[i];
    }
#pragma unroll
    for (int i = 0; i < 8; ++i) {
      int p = tid + i * 256;
      int kk = p >> 6, n = (p & 63) * 2;
      BsT[n][kk]     = bLo[i];
      BsT[n + 1][kk] = bHi[i];
    }
  };

  stageLoad(0);
  for (int k0 = 0; k0 < K; k0 += 32) {
    __syncthreads();                        // LDS free from previous compute
    stageStore();
    __syncthreads();
    if (k0 + 32 < K) stageLoad(k0 + 32);    // overlap next loads with WMMA burst
    FragA a, b0, b1, b2, b3;
#pragma unroll
    for (int j = 0; j < 8; ++j) {
      int kb = ((j & 4) << 2) + hh * 8 + (j & 3) * 2;   // ISA A/B K packing
      a.u[j]  = *(const unsigned*)&As[wrow * 16 + mloc][kb];
      b0.u[j] = *(const unsigned*)&BsT[wcol * 64 + mloc][kb];
      b1.u[j] = *(const unsigned*)&BsT[wcol * 64 + 16 + mloc][kb];
      b2.u[j] = *(const unsigned*)&BsT[wcol * 64 + 32 + mloc][kb];
      b3.u[j] = *(const unsigned*)&BsT[wcol * 64 + 48 + mloc][kb];
    }
    acc[0] = wmma_bf16(a, b0, acc[0]);
    acc[1] = wmma_bf16(a, b1, acc[1]);
    acc[2] = wmma_bf16(a, b2, acc[2]);
    acc[3] = wmma_bf16(a, b3, acc[3]);
  }

  const int gmb = m0 + wrow * 16 + hh * 8;  // C layout: row = base + j
#pragma unroll
  for (int t = 0; t < 4; ++t) {
    int gn = n0 + wcol * 64 + t * 16 + mloc;
    if (gn >= N) continue;
    float bv = bias ? bias[gn] : 0.0f;
#pragma unroll
    for (int j = 0; j < 8; ++j) {
      int gm = gmb + j;
      if (gm >= M) continue;
      float r = acc[t][j] + bv;
      if (GELU_FLAG) r = 0.5f * r * (1.0f + erff(r * 0.70710678118654752f));
      size_t o = (size_t)gm * N + gn;
      if (OUT_MODE == 0)      ((float*)Out)[o] = r;
      else if (OUT_MODE == 1) ((float*)Out)[o] += r;
      else ((unsigned short*)Out)[o] = f32_to_bf16(r);
    }
  }
}

// ---------------------------------------------------------------------------
// Flash attention (causal). Block = 128 query rows (8 waves x 16), one (b,h).
// K tile staged via GLOBAL_LOAD_ASYNC_TO_LDS_B128 (ASYNCcnt path); V tile
// transposed through VGPRs. Online softmax; all matmuls WMMA.
// ---------------------------------------------------------------------------
__global__ __launch_bounds__(256) void attn_k(const unsigned short* __restrict__ Q,
                                              const unsigned short* __restrict__ Kmat,
                                              const unsigned short* __restrict__ Vmat,
                                              unsigned short* __restrict__ Y) {
  const int b = blockIdx.z, h = blockIdx.y, qt = blockIdx.x;
  const int tid = threadIdx.x, lane = tid & 31, wave = tid >> 5;
  const int mloc = lane & 15, hh = lane >> 4;
  const int qbase = qt * 128 + wave * 16;

  __shared__ unsigned short Kt[64][64];        // [key][d]   (async-filled)
  __shared__ unsigned short Vt[64][64];        // [d][key]   (transposed)
  __shared__ unsigned short Ps[8][16][64];     // per-wave P scratch

  // Q fragments (16 rows x 64 d = two 16x32 A-frags), loaded once
  FragA aq[2];
#pragma unroll
  for (int ks = 0; ks < 2; ++ks)
#pragma unroll
    for (int j = 0; j < 8; ++j) {
      int kb = ((j & 4) << 2) + hh * 8 + (j & 3) * 2 + ks * 32;
      aq[ks].u[j] = *(const unsigned*)(Q + ((size_t)(b * TN + qbase + mloc) * CN + h * DN + kb));
    }

  v8f O[4] = {v8f{}, v8f{}, v8f{}, v8f{}};
  float mrow[8], lrow[8];
#pragma unroll
  for (int j = 0; j < 8; ++j) { mrow[j] = -1e30f; lrow[j] = 0.f; }

  const int ntiles = 2 * qt + 2;               // causal: keys up to block end
  for (int jt = 0; jt < ntiles; ++jt) {
    const int keybase = jt * 64;
    __syncthreads();
    // --- K tile: async DMA global -> LDS, 16B per lane-op, 2 ops/thread ---
#pragma unroll
    for (int i = 0; i < 2; ++i) {
      int e   = tid + i * 256;                 // 512 x 16-byte chunks = 8KB
      int key = e >> 3;
      int ch  = e & 7;                         // 8 chunks per 128B row
      unsigned long long gaddr = (unsigned long long)(uintptr_t)
          (Kmat + (size_t)(b * TN + keybase + key) * CN + h * DN + ch * 8);
      unsigned lds = (unsigned)(uintptr_t)&Kt[key][ch * 8];
      asm volatile("global_load_async_to_lds_b128 %0, %1, off"
                   :: "v"(lds), "v"(gaddr) : "memory");
    }
    // --- V tile: load + transpose into LDS ---
#pragma unroll
    for (int i = 0; i < 8; ++i) {
      int e   = tid + i * 256;
      int key = e >> 5;
      int dp  = (e & 31) * 2;
      unsigned vv = *(const unsigned*)(Vmat + ((size_t)(b * TN + keybase + key) * CN + h * DN + dp));
      Vt[dp][key]     = (unsigned short)(vv & 0xffffu);
      Vt[dp + 1][key] = (unsigned short)(vv >> 16);
    }
    asm volatile("s_wait_asynccnt 0" ::: "memory");
    __syncthreads();

    // S = Q @ K^T * scale, causal mask with -10000 (matches reference)
    v8f S[4];
#pragma unroll
    for (int nt = 0; nt < 4; ++nt) {
      v8f s = {};
#pragma unroll
      for (int ks = 0; ks < 2; ++ks) {
        FragA bk;
#pragma unroll
        for (int j = 0; j < 8; ++j) {
          int kb = ((j & 4) << 2) + hh * 8 + (j & 3) * 2 + ks * 32;
          bk.u[j] = *(const unsigned*)&Kt[nt * 16 + mloc][kb];
        }
        s = wmma_bf16(aq[ks], bk, s);
      }
      int keyi = keybase + nt * 16 + mloc;
#pragma unroll
      for (int j = 0; j < 8; ++j) {
        int rowi = qbase + j + 8 * hh;
        S[nt][j] = (keyi <= rowi) ? s[j] * 0.125f : -10000.0f;
      }
    }

    // online softmax per row (row lives across 16 lanes of a half-wave)
#pragma unroll
    for (int j = 0; j < 8; ++j) {
      float v = fmaxf(fmaxf(S[0][j], S[1][j]), fmaxf(S[2][j], S[3][j]));
#pragma unroll
      for (int o = 8; o; o >>= 1) v = fmaxf(v, __shfl_xor(v, o, 32));
      float mnew  = fmaxf(mrow[j], v);
      float alpha = expf(mrow[j] - mnew);
      float rsum = 0.f;
#pragma unroll
      for (int nt = 0; nt < 4; ++nt) {
        float p = expf(S[nt][j] - mnew);
        S[nt][j] = p;
        rsum += p;
      }
#pragma unroll
      for (int o = 8; o; o >>= 1) rsum += __shfl_xor(rsum, o, 32);
      lrow[j] = lrow[j] * alpha + rsum;
      mrow[j] = mnew;
#pragma unroll
      for (int dt = 0; dt < 4; ++dt) O[dt][j] *= alpha;
    }

    // C-layout P -> LDS (A-layout source) as bf16
#pragma unroll
    for (int nt = 0; nt < 4; ++nt)
#pragma unroll
      for (int j = 0; j < 8; ++j)
        Ps[wave][j + 8 * hh][nt * 16 + mloc] = f32_to_bf16(S[nt][j]);

    // O += P @ V
#pragma unroll
    for (int ks2 = 0; ks2 < 2; ++ks2) {
      FragA ap;
#pragma unroll
      for (int j = 0; j < 8; ++j) {
        int kb = ((j & 4) << 2) + hh * 8 + (j & 3) * 2 + ks2 * 32;
        ap.u[j] = *(const unsigned*)&Ps[wave][mloc][kb];
      }
#pragma unroll
      for (int dt = 0; dt < 4; ++dt) {
        FragA bv;
#pragma unroll
        for (int j = 0; j < 8; ++j) {
          int kb = ((j & 4) << 2) + hh * 8 + (j & 3) * 2 + ks2 * 32;
          bv.u[j] = *(const unsigned*)&Vt[dt * 16 + mloc][kb];
        }
        O[dt] = wmma_bf16(ap, bv, O[dt]);
      }
    }
  }

  // normalize and store y (bf16)
#pragma unroll
  for (int j = 0; j < 8; ++j) {
    float inv = 1.0f / lrow[j];
    size_t rowoff = (size_t)(b * TN + qbase + j + 8 * hh) * CN + h * DN;
#pragma unroll
    for (int dt = 0; dt < 4; ++dt)
      Y[rowoff + dt * 16 + mloc] = f32_to_bf16(O[dt][j] * inv);
  }
}

// ---------------------------------------------------------------------------
// Orchestration
// ---------------------------------------------------------------------------
extern "C" void kernel_launch(void* const* d_in, const int* in_sizes, int n_in,
                              void* d_out, int out_size, void* d_ws, size_t ws_size,
                              hipStream_t stream) {
  (void)in_sizes; (void)n_in; (void)out_size; (void)ws_size;
  const int*   idx   = (const int*)d_in[0];
  const float* tok   = (const float*)d_in[1];
  const float* pos   = (const float*)d_in[2];
  const float* ln1g  = (const float*)d_in[3];
  const float* ln1b  = (const float*)d_in[4];
  const float* Wq    = (const float*)d_in[5];
  const float* bq    = (const float*)d_in[6];
  const float* Wk    = (const float*)d_in[7];
  const float* bk_   = (const float*)d_in[8];
  const float* Wv    = (const float*)d_in[9];
  const float* bv_   = (const float*)d_in[10];
  const float* Wp    = (const float*)d_in[11];
  const float* bp    = (const float*)d_in[12];
  const float* ln2g  = (const float*)d_in[13];
  const float* ln2b  = (const float*)d_in[14];
  const float* Wf1   = (const float*)d_in[15];
  const float* bf1   = (const float*)d_in[16];
  const float* Wf2   = (const float*)d_in[17];
  const float* bf2   = (const float*)d_in[18];
  const float* lnfg  = (const float*)d_in[19];
  const float* lnfb  = (const float*)d_in[20];
  const float* headW = (const float*)d_in[21];

  char* base = (char*)d_ws;
  size_t off = 0;
  auto take = [&](size_t bytes) -> char* {
    char* p = base + off;
    off = (off + bytes + 255) & ~(size_t)255;
    return p;
  };
  float*          x   = (float*)take((size_t)MN * CN * 4);
  unsigned short* hb  = (unsigned short*)take((size_t)MN * CN * 2);
  unsigned short* qb  = (unsigned short*)take((size_t)MN * CN * 2);
  unsigned short* kb2 = (unsigned short*)take((size_t)MN * CN * 2);
  unsigned short* vb2 = (unsigned short*)take((size_t)MN * CN * 2);
  unsigned short* yb  = (unsigned short*)take((size_t)MN * CN * 2);
  unsigned short* f1  = (unsigned short*)take((size_t)MN * FFN * 2);
  unsigned short* wbuf = (unsigned short*)take((size_t)CN * (size_t)VOCN * 2); // weight staging
  const size_t cc = (size_t)CN * CN;
  const size_t cf = (size_t)CN * FFN;
  unsigned short* wq_s = wbuf;
  unsigned short* wk_s = wbuf + cc;
  unsigned short* wv_s = wbuf + 2 * cc;
  unsigned short* wp_s = wbuf + 3 * cc;
  unsigned short* w1_s = wbuf + 4 * cc;   // C x 4C
  unsigned short* w2_s = wbuf + 8 * cc;   // 4C x C

  dim3 blk(256);
  int tot = MN * CN;
  hipLaunchKernelGGL(embed_k, dim3((tot + 255) / 256), blk, 0, stream, idx, tok, pos, x, tot);

  dim3 gC((CN + 127) / 128, MN / 64);
  dim3 gF((FFN + 127) / 128, MN / 64);
  dim3 gHead((VOCN + 127) / 128, MN / 64);
  dim3 gAttn(TN / 128, HN, BN);

  for (int l = 0; l < NLAYER; ++l) {
    hipLaunchKernelGGL(cvt_k, dim3(((int)cc + 255) / 256), blk, 0, stream, Wq + l * cc, wq_s, (int)cc);
    hipLaunchKernelGGL(cvt_k, dim3(((int)cc + 255) / 256), blk, 0, stream, Wk + l * cc, wk_s, (int)cc);
    hipLaunchKernelGGL(cvt_k, dim3(((int)cc + 255) / 256), blk, 0, stream, Wv + l * cc, wv_s, (int)cc);
    hipLaunchKernelGGL(cvt_k, dim3(((int)cc + 255) / 256), blk, 0, stream, Wp + l * cc, wp_s, (int)cc);
    hipLaunchKernelGGL(cvt_k, dim3(((int)cf + 255) / 256), blk, 0, stream, Wf1 + l * cf, w1_s, (int)cf);
    hipLaunchKernelGGL(cvt_k, dim3(((int)cf + 255) / 256), blk, 0, stream, Wf2 + l * cf, w2_s, (int)cf);

    hipLaunchKernelGGL(layernorm_k, dim3(MN), blk, 0, stream, x, ln1g + l * CN, ln1b + l * CN, hb, CN);
    hipLaunchKernelGGL((gemm_wmma_k<2, false>), gC, blk, 0, stream, hb, wq_s, bq + l * CN, (void*)qb,  MN, CN, CN);
    hipLaunchKernelGGL((gemm_wmma_k<2, false>), gC, blk, 0, stream, hb, wk_s, bk_ + l * CN, (void*)kb2, MN, CN, CN);
    hipLaunchKernelGGL((gemm_wmma_k<2, false>), gC, blk, 0, stream, hb, wv_s, bv_ + l * CN, (void*)vb2, MN, CN, CN);
    hipLaunchKernelGGL(attn_k, gAttn, blk, 0, stream, qb, kb2, vb2, yb);
    hipLaunchKernelGGL((gemm_wmma_k<1, false>), gC, blk, 0, stream, yb, wp_s, bp + l * CN, (void*)x, MN, CN, CN);
    hipLaunchKernelGGL(layernorm_k, dim3(MN), blk, 0, stream, x, ln2g + l * CN, ln2b + l * CN, hb, CN);
    hipLaunchKernelGGL((gemm_wmma_k<2, true>),  gF, blk, 0, stream, hb, w1_s, bf1 + (size_t)l * FFN, (void*)f1, MN, FFN, CN);
    hipLaunchKernelGGL((gemm_wmma_k<1, false>), gC, blk, 0, stream, f1, w2_s, bf2 + l * CN, (void*)x, MN, CN, FFN);
  }

  hipLaunchKernelGGL(layernorm_k, dim3(MN), blk, 0, stream, x, lnfg, lnfb, hb, CN);
  int nhead = CN * VOCN;
  hipLaunchKernelGGL(cvt_k, dim3((nhead + 255) / 256), blk, 0, stream, headW, wbuf, nhead);
  hipLaunchKernelGGL((gemm_wmma_k<0, false>), gHead, blk, 0, stream, hb, wbuf, (const float*)nullptr,
                     d_out, MN, VOCN, CN);
}
